// LinearCrossAttention_91001767068155
// MI455X (gfx1250) — matrix operand order repaired and checked by reading, not compiled, not run on hardware
//
#include <hip/hip_runtime.h>
#include <hip/hip_bf16.h>
#include <math.h>
#include <stdint.h>

// ---------------------------------------------------------------------------
// CDNA5 (gfx1250) fused linear cross-attention.
// B=8, C=512, T=8192, TOK=256, CD=768, H=8, D=64.
// ---------------------------------------------------------------------------

typedef __bf16 bf16;
typedef __attribute__((ext_vector_type(16))) __bf16 v16bf;
typedef __attribute__((ext_vector_type(8)))  __bf16 v8bf;
typedef __attribute__((ext_vector_type(8)))  float  v8f;

#define WMMA_BF16(a, b, c) \
    __builtin_amdgcn_wmma_f32_16x16x32_bf16(false, (a), false, (b), (short)0, (c), false, false)

__device__ __forceinline__ v8f vzero8() {
    v8f z;
#pragma unroll
    for (int i = 0; i < 8; ++i) z[i] = 0.0f;
    return z;
}

// A fragment (16x32 bf16, M x K) from a row-major bf16 buffer (16B-aligned rows).
// Contiguous K order (matches B loaders): lane pair (m, m+16): a[j] = K = 16*half + j, j = 0..15.
// Two 16-byte vector loads -> ds_load_b128 / global_load_b128.
__device__ __forceinline__ v16bf load_a_bf16(const bf16* base, int ld, int lane) {
    const int m = lane & 15, half = lane >> 4;
    const bf16* p = base + m * ld + half * 16;
    const v8bf lo = *(const v8bf*)p;
    const v8bf hi = *(const v8bf*)(p + 8);
    v16bf a;
#pragma unroll
    for (int j = 0; j < 8; ++j) { a[j] = lo[j]; a[j + 8] = hi[j]; }
    return a;
}

// B fragment (32x16, K x N) from pre-converted bf16 weights, K contiguous in memory:
// caller passes p = &W[(n0 + (lane&15))*ldw + k0 + 16*(lane>>4)]; 32 contiguous bytes.
__device__ __forceinline__ v16bf load_b_bf16_contig(const bf16* p) {
    return *(const v16bf*)p;
}

// B fragment where K advances with a stride (bf16 attn matrix, N contiguous):
// caller passes p = &A[(k0 + 16*(lane>>4))*stride + n0 + (lane&15)].
__device__ __forceinline__ v16bf load_b_kmajor_bf16(const bf16* p, int stride) {
    v16bf b;
#pragma unroll
    for (int j = 0; j < 16; ++j) b[j] = p[j * stride];
    return b;
}

// ---------------------------------------------------------------------------
// Kernel W: one-shot fp32 -> bf16 weight conversion.
// ---------------------------------------------------------------------------
__global__ __launch_bounds__(256) void cvt_f32_bf16_kernel(
    const float* __restrict__ src, bf16* __restrict__ dst, int n) {
    const int i = blockIdx.x * 256 + threadIdx.x;
    if (i < n) dst[i] = (bf16)src[i];
}

// ---------------------------------------------------------------------------
// Kernel A: layernorm(context) -> bf16  (rows = B*TOK = 2048, width CD = 768)
// ---------------------------------------------------------------------------
__global__ __launch_bounds__(256) void ctx_ln_kernel(
    const float* __restrict__ ctx, const float* __restrict__ g,
    const float* __restrict__ bta, bf16* __restrict__ outc) {
    const int row  = blockIdx.x * 8 + (threadIdx.x >> 5);
    const int lane = threadIdx.x & 31;
    const float* p = ctx + (size_t)row * 768;
    float s = 0.f, sq = 0.f;
    for (int c = lane; c < 768; c += 32) { float v = p[c]; s += v; sq += v * v; }
#pragma unroll
    for (int m = 1; m < 32; m <<= 1) { s += __shfl_xor(s, m, 32); sq += __shfl_xor(sq, m, 32); }
    const float mean = s * (1.f / 768.f);
    const float var  = sq * (1.f / 768.f) - mean * mean;
    const float rstd = rsqrtf(var + 1e-5f);
    bf16* o = outc + (size_t)row * 768;
    for (int c = lane; c < 768; c += 32)
        o[c] = (bf16)((p[c] - mean) * rstd * g[c] + bta[c]);
}

// ---------------------------------------------------------------------------
// Kernel B: k/v projections via WMMA (bf16 weights). One wave per 16x16 tile.
// grid: x = (B*TOK)/16 = 128, y = C/16 = 32, z = 2 (0 -> k, 1 -> v)
// ---------------------------------------------------------------------------
__global__ __launch_bounds__(32) void kv_proj_kernel(
    const bf16* __restrict__ ctxn,
    const bf16* __restrict__ Wkb, const float* __restrict__ bk,
    const bf16* __restrict__ Wvb, const float* __restrict__ bv,
    float* __restrict__ kout, float* __restrict__ vout) {
    const bf16*  W    = blockIdx.z ? Wvb : Wkb;
    const float* bias = blockIdx.z ? bv : bk;
    float*       dst  = blockIdx.z ? vout : kout;
    const int r0 = blockIdx.x * 16;
    const int n0 = blockIdx.y * 16;
    const int lane = threadIdx.x;
    const int half = lane >> 4, nl = lane & 15;

    v8f acc = vzero8();
    for (int k0 = 0; k0 < 768; k0 += 32) {
        v16bf a = load_a_bf16(ctxn + (size_t)r0 * 768 + k0, 768, lane);
        v16bf b = load_b_bf16_contig(W + (size_t)(n0 + nl) * 768 + k0 + 16 * half);
        acc = WMMA_BF16(a, b, acc);
    }
    const float bb = bias[n0 + nl];
#pragma unroll
    for (int i = 0; i < 8; ++i) {
        const int row = r0 + i + 8 * half;
        dst[(size_t)row * 512 + n0 + nl] = acc[i] + bb;
    }
}

// ---------------------------------------------------------------------------
// Kernel C: mask, softmax over TOK, attn[b,h] = k_sm^T @ v_m  (64x64, bf16 out)
// grid = B*H = 64 blocks, 256 threads. Dynamic LDS: 2*256*64*4 + 256 bytes.
// ---------------------------------------------------------------------------
__global__ __launch_bounds__(256) void attn_mat_kernel(
    const float* __restrict__ kin, const float* __restrict__ vin,
    const unsigned char* __restrict__ mask, bf16* __restrict__ attn) {
    extern __shared__ char smem[];
    float* kl  = (float*)smem;                 // 256*64 fp32
    float* vl  = kl + 256 * 64;                // 256*64 fp32
    float* inv = vl + 256 * 64;                // 64 fp32

    const int b = blockIdx.x >> 3, h = blockIdx.x & 7;
    const int tid = threadIdx.x;

    for (int idx = tid; idx < 256 * 64; idx += 256) {
        const int n = idx >> 6, d = idx & 63;
        const bool on = mask[b * 256 + n] != 0;
        const size_t src = ((size_t)(b * 256 + n)) * 512 + h * 64 + d;
        kl[idx] = on ? kin[src] : -3.402823466e38f;
        vl[idx] = on ? vin[src] : 0.0f;
    }
    __syncthreads();

    if (tid < 64) {
        const int d = tid;
        float m = -3.402823466e38f;
        for (int n = 0; n < 256; ++n) m = fmaxf(m, kl[n * 64 + d]);
        float s = 0.f;
        for (int n = 0; n < 256; ++n) {
            float e = __expf(kl[n * 64 + d] - m);
            kl[n * 64 + d] = e;
            s += e;
        }
        inv[d] = 1.0f / s;
    }
    __syncthreads();

    const int d  = tid >> 2;
    const int e0 = (tid & 3) * 16;
    float acc[16];
#pragma unroll
    for (int j = 0; j < 16; ++j) acc[j] = 0.f;
    for (int n = 0; n < 256; ++n) {
        const float kv = kl[n * 64 + d];
#pragma unroll
        for (int j = 0; j < 16; ++j) acc[j] += kv * vl[n * 64 + e0 + j];
    }
    const float sc = inv[d];
    bf16* dstp = attn + ((size_t)(b * 8 + h)) * 4096 + d * 64 + e0;
#pragma unroll
    for (int j = 0; j < 16; ++j) dstp[j] = (bf16)(acc[j] * sc);
}

// ---------------------------------------------------------------------------
// Kernel D (main, fused): per (b, 32-row T tile):
//   stage x tile -> LN -> bf16 -> WMMA GEMM1 (q = hn @ Wq^T + bq, bf16 weights)
//   -> per-head softmax over D (in-register, 16-lane shuffles)
//   -> WMMA GEMM2 (q_sm @ attn[h], bf16 attn) -> residual -> coalesced store.
// Wave w owns head w (output columns [64w, 64w+64)).
// grid = (T/32, B) = (256, 8), 256 threads (8 waves).
// ---------------------------------------------------------------------------
#define TT   32
#define XLD  513   // fp32 tile leading dim (stride%64==1 -> conflict-free column access)
#define QLD  520   // bf16 tile leading dim (16B-aligned rows)

__global__ __launch_bounds__(256) void fused_q_attn_kernel(
    const float* __restrict__ x,  const float* __restrict__ ln_g,
    const float* __restrict__ ln_b, const bf16* __restrict__ Wqb,
    const float* __restrict__ bq, const bf16* __restrict__ attn,
    float* __restrict__ out) {
    extern __shared__ char smem[];
    float* xt = (float*)smem;                       // TT*XLD fp32 = 65,664 B
    bf16*  qt = (bf16*)(smem + TT * XLD * 4);       // TT*QLD bf16 = 33,280 B

    const int b   = blockIdx.y;
    const int t0  = blockIdx.x * TT;
    const int tid = threadIdx.x;
    const int lane = tid & 31, wave = tid >> 5;
    const int half = lane >> 4, nl = lane & 15;
    const int T = 8192, C = 512;

    // --- 1) stage x tile: hidden[t][c] = x[b, c, t0+t] (coalesced along t) ---
    {
        const int tl = tid & 31;
        for (int c = tid >> 5; c < C; c += 8)
            xt[tl * XLD + c] = x[((size_t)b * C + c) * T + t0 + tl];
    }
    __syncthreads();

    // --- 2) layernorm rows -> bf16 tile (8 lanes per row) ---
    {
        const int r = tid >> 3, s = tid & 7;
        float sum = 0.f, sumsq = 0.f;
        for (int c = s; c < C; c += 8) {
            const float v = xt[r * XLD + c];
            sum += v; sumsq += v * v;
        }
#pragma unroll
        for (int m = 1; m < 8; m <<= 1) {
            sum   += __shfl_xor(sum, m, 32);
            sumsq += __shfl_xor(sumsq, m, 32);
        }
        const float mean = sum * (1.f / 512.f);
        const float var  = sumsq * (1.f / 512.f) - mean * mean;
        const float rstd = rsqrtf(var + 1e-5f);
        for (int c = s; c < C; c += 8) {
            const float v = (xt[r * XLD + c] - mean) * rstd * ln_g[c] + ln_b[c];
            qt[r * QLD + c] = (bf16)v;
        }
    }
    __syncthreads();

    // --- 3) GEMM1: wave w computes q columns [64w, 64w+64), rows 0..31 ---
    const int h = wave;
    const int nbase = h * 64;
    v8f acc[2][4];
#pragma unroll
    for (int mt = 0; mt < 2; ++mt)
#pragma unroll
        for (int f = 0; f < 4; ++f) acc[mt][f] = vzero8();

    // bf16 Wq row base for this lane's column group
    const bf16* wrow = Wqb + (size_t)(nbase + nl) * 512 + 16 * half;

    for (int k0 = 0; k0 < 512; k0 += 32) {
        v16bf a0 = load_a_bf16(qt + 0 * 16 * QLD + k0, QLD, lane);
        v16bf a1 = load_a_bf16(qt + 1 * 16 * QLD + k0, QLD, lane);
        // prefetch next K-step's weight line for f=0 (gfx1250 global_prefetch_b8)
        if (k0 + 32 < 512) __builtin_prefetch(wrow + k0 + 32, 0, 0);
#pragma unroll
        for (int f = 0; f < 4; ++f) {
            v16bf bfrag = load_b_bf16_contig(wrow + (size_t)f * 16 * 512 + k0);
            acc[0][f] = WMMA_BF16(a0, bfrag, acc[0][f]);
            acc[1][f] = WMMA_BF16(a1, bfrag, acc[1][f]);
        }
    }

    // --- 4) bias + softmax over the head's 64 columns (per row) ---
#pragma unroll
    for (int f = 0; f < 4; ++f) {
        const float bqv = bq[nbase + f * 16 + nl];
#pragma unroll
        for (int i = 0; i < 8; ++i) { acc[0][f][i] += bqv; acc[1][f][i] += bqv; }
    }
#pragma unroll
    for (int mt = 0; mt < 2; ++mt) {
#pragma unroll
        for (int i = 0; i < 8; ++i) {
            float mx = fmaxf(fmaxf(acc[mt][0][i], acc[mt][1][i]),
                             fmaxf(acc[mt][2][i], acc[mt][3][i]));
#pragma unroll
            for (int m = 1; m < 16; m <<= 1) mx = fmaxf(mx, __shfl_xor(mx, m, 32));
            float s = 0.f;
#pragma unroll
            for (int f = 0; f < 4; ++f) {
                const float e = __expf(acc[mt][f][i] - mx);
                acc[mt][f][i] = e;
                s += e;
            }
#pragma unroll
            for (int m = 1; m < 16; m <<= 1) s += __shfl_xor(s, m, 32);
            const float invs = 1.0f / s;
#pragma unroll
            for (int f = 0; f < 4; ++f) acc[mt][f][i] *= invs;
        }
    }

    // --- 5) write softmaxed q back to LDS (bf16), reuse qt ---
    __syncthreads();   // all waves finished reading qt for GEMM1
#pragma unroll
    for (int mt = 0; mt < 2; ++mt)
#pragma unroll
        for (int f = 0; f < 4; ++f)
#pragma unroll
            for (int i = 0; i < 8; ++i) {
                const int row = mt * 16 + i + 8 * half;
                const int col = nbase + f * 16 + nl;
                qt[row * QLD + col] = (bf16)acc[mt][f][i];
            }
    __syncthreads();

    // --- 6) GEMM2: out_head = q_sm[:, head] @ attn[b,h] (64x64 bf16) ---
    const bf16* ah = attn + ((size_t)(b * 8 + h)) * 4096;
    v8f o2[2][4];
#pragma unroll
    for (int mt = 0; mt < 2; ++mt)
#pragma unroll
        for (int f = 0; f < 4; ++f) o2[mt][f] = vzero8();

#pragma unroll
    for (int kk = 0; kk < 64; kk += 32) {
        v16bf a0 = load_a_bf16(qt + 0 * 16 * QLD + nbase + kk, QLD, lane);
        v16bf a1 = load_a_bf16(qt + 1 * 16 * QLD + nbase + kk, QLD, lane);
#pragma unroll
        for (int f = 0; f < 4; ++f) {
            const bf16* bp = ah + (kk + 16 * half) * 64 + f * 16 + nl;
            v16bf bfrag = load_b_kmajor_bf16(bp, 64);
            o2[0][f] = WMMA_BF16(a0, bfrag, o2[0][f]);
            o2[1][f] = WMMA_BF16(a1, bfrag, o2[1][f]);
        }
    }

    // --- 7) residual into xt (each (row,col) owned by exactly one lane) ---
#pragma unroll
    for (int mt = 0; mt < 2; ++mt)
#pragma unroll
        for (int f = 0; f < 4; ++f)
#pragma unroll
            for (int i = 0; i < 8; ++i) {
                const int row = mt * 16 + i + 8 * half;
                const int col = nbase + f * 16 + nl;
                xt[row * XLD + col] += o2[mt][f][i];
            }
    __syncthreads();

    // --- 8) coalesced store back to (B, C, T) layout ---
    {
        const int tl = tid & 31;
        for (int c = tid >> 5; c < C; c += 8)
            out[((size_t)b * C + c) * T + t0 + tl] = xt[tl * XLD + c];
    }
}

// ---------------------------------------------------------------------------
// Host launcher
// ---------------------------------------------------------------------------
extern "C" void kernel_launch(void* const* d_in, const int* in_sizes, int n_in,
                              void* d_out, int out_size, void* d_ws, size_t ws_size,
                              hipStream_t stream) {
    (void)in_sizes; (void)n_in; (void)out_size; (void)ws_size;
    const float*         x            = (const float*)d_in[0];
    const float*         context      = (const float*)d_in[1];
    const unsigned char* context_mask = (const unsigned char*)d_in[2]; // bool array
    const float*         ln_g         = (const float*)d_in[3];
    const float*         ln_b         = (const float*)d_in[4];
    const float*         cln_g        = (const float*)d_in[5];
    const float*         cln_b        = (const float*)d_in[6];
    const float*         Wq           = (const float*)d_in[7];
    const float*         bq           = (const float*)d_in[8];
    const float*         Wk           = (const float*)d_in[9];
    const float*         bk           = (const float*)d_in[10];
    const float*         Wv           = (const float*)d_in[11];
    const float*         bv           = (const float*)d_in[12];
    float*               out          = (float*)d_out;

    // Workspace layout (~14.2 MB)
    char* ws = (char*)d_ws;
    size_t off = 0;
    bf16*  ws_ctx  = (bf16*)(ws + off); off += (size_t)2048 * 768 * 2;   // 3,145,728
    float* ws_k    = (float*)(ws + off); off += (size_t)2048 * 512 * 4;  // 4,194,304
    float* ws_v    = (float*)(ws + off); off += (size_t)2048 * 512 * 4;  // 4,194,304
    bf16*  ws_attn = (bf16*)(ws + off); off += (size_t)64 * 4096 * 2;    //   524,288
    bf16*  ws_wq   = (bf16*)(ws + off); off += (size_t)512 * 512 * 2;    //   524,288
    bf16*  ws_wk   = (bf16*)(ws + off); off += (size_t)512 * 768 * 2;    //   786,432
    bf16*  ws_wv   = (bf16*)(ws + off); off += (size_t)512 * 768 * 2;    //   786,432

    // W) one-shot weight conversion fp32 -> bf16
    cvt_f32_bf16_kernel<<<(512 * 512 + 255) / 256, 256, 0, stream>>>(Wq, ws_wq, 512 * 512);
    cvt_f32_bf16_kernel<<<(512 * 768 + 255) / 256, 256, 0, stream>>>(Wk, ws_wk, 512 * 768);
    cvt_f32_bf16_kernel<<<(512 * 768 + 255) / 256, 256, 0, stream>>>(Wv, ws_wv, 512 * 768);

    // A) LN(context) -> bf16
    ctx_ln_kernel<<<256, 256, 0, stream>>>(context, cln_g, cln_b, ws_ctx);

    // B) k/v projections (WMMA, bf16 operands)
    kv_proj_kernel<<<dim3(128, 32, 2), 32, 0, stream>>>(ws_ctx, ws_wk, bk, ws_wv, bv, ws_k, ws_v);

    // C) mask + softmax(TOK) + attn = k^T v  -> bf16  (128 KB + 256 B dynamic LDS)
    attn_mat_kernel<<<64, 256, (2 * 256 * 64 + 64) * sizeof(float), stream>>>(
        ws_k, ws_v, context_mask, ws_attn);

    // D) fused LN -> GEMM1 -> softmax(D) -> GEMM2 -> residual (~96.6 KB dynamic LDS)
    const size_t smemD = (size_t)TT * XLD * 4 + (size_t)TT * QLD * 2;
    fused_q_attn_kernel<<<dim3(8192 / TT, 8), 256, smemD, stream>>>(
        x, ln_g, ln_b, ws_wq, bq, ws_attn, out);
}
